// MoTMLP_54700703482360
// MI455X (gfx1250) — compile-verified, matrix operand order checked by reading, not run on hardware
//
#include <hip/hip_runtime.h>
#include <hip/hip_bf16.h>

// ---------------- problem constants (match reference) ----------------
constexpr int Bc = 4;
constexpr int Qc = 4096;
constexpr int Hc = 1024;   // hidden
constexpr int Ic = 4096;   // intermediate
constexpr int Ec = 8;      // experts
constexpr int Tc = Bc * Qc;           // 16384 tokens
constexpr float EPSc = 1e-5f;

// ---------------- WMMA vector types (gfx1250, wave32) ----------------
typedef __attribute__((ext_vector_type(16))) _Float16 v16h;
typedef __attribute__((ext_vector_type(8)))  _Float16 v8h;
typedef __attribute__((ext_vector_type(8)))  float    v8f;

// GEMM tiling
constexpr int BM = 128;
constexpr int BN = 128;
constexpr int BK = 32;
constexpr int LDT = 40;    // padded LDS stride in halves (80B: 16B-aligned, 20-bank stride)
constexpr int TILE_ELEMS = BM * LDT;

// ---------------------------------------------------------------------
// helpers
// ---------------------------------------------------------------------
__device__ __forceinline__ float gelu_tanh(float x) {
    float u = 0.7978845608028654f * (x + 0.044715f * x * x * x);
    return 0.5f * x * (1.0f + tanhf(u));
}

__device__ __forceinline__ unsigned int lds_off_u32(const void* p) {
    // ISA 10.2: flat->LDS mapping truncates to addr[31:0]; generic LDS
    // pointers carry the LDS byte offset in their low 32 bits.
    return (unsigned int)(unsigned long long)p;
}

// async copy: 16 bytes global -> LDS, tracked by ASYNCcnt (CDNA5 path)
__device__ __forceinline__ void async_b128(unsigned int lds_byte_off, const void* gptr) {
    asm volatile("global_load_async_to_lds_b128 %0, %1, off"
                 :: "v"(lds_byte_off), "v"(gptr) : "memory");
}

__device__ __forceinline__ void wait_async0() {
    asm volatile("s_wait_asynccnt 0x0" ::: "memory");
}

// Stage a 128x32 f16 tile (row-major source, K-contiguous) into LDS with
// padded stride LDT, via per-lane async b128 chunks. 256 threads x 2 chunks.
__device__ __forceinline__ void stage_tile_async(_Float16* lds_tile,
                                                 const _Float16* __restrict__ gbase,
                                                 int ld, int row0, int k0, int tid) {
    const unsigned int lbase = lds_off_u32(lds_tile);
#pragma unroll
    for (int p = 0; p < 2; ++p) {
        const int c   = tid + p * 256;     // 0..511 : 128 rows x 4 chunks
        const int row = c >> 2;
        const int ch  = c & 3;
        const unsigned int loff = lbase + (unsigned int)(row * LDT + ch * 8) * 2u;
        const void* g = (const void*)(gbase + (size_t)(row0 + row) * ld + k0 + ch * 8);
        async_b128(loff, g);
    }
}

// Load one 16x32 f16 fragment (ISA 7.12.2 layout) from padded LDS tile.
__device__ __forceinline__ v16h load_frag(const _Float16* tile, int row16, int lane) {
    const int row  = row16 + (lane & 15);
    const int koff = (lane >> 4) * 8;
    const v8h lo = *(const v8h*)(tile + row * LDT + koff);
    const v8h hi = *(const v8h*)(tile + row * LDT + koff + 16);
    v16h f;
#pragma unroll
    for (int i = 0; i < 8; ++i) { f[i] = lo[i]; f[8 + i] = hi[i]; }
    return f;
}

// ---------------------------------------------------------------------
// Kernel 1: router (softmax + top2 renorm) + LayerNorm, one wave / token.
// ---------------------------------------------------------------------
__global__ __launch_bounds__(256)
void router_ln_kernel(const float* __restrict__ x,
                      const float* __restrict__ rw,   // [H,E]
                      const float* __restrict__ rb,   // [E]
                      float* __restrict__ score_out,  // [T,E]
                      float* __restrict__ combine,    // [T,E]
                      _Float16* __restrict__ xn)      // [T,H]
{
    const int wave = (blockIdx.x * blockDim.x + threadIdx.x) >> 5;
    const int lane = threadIdx.x & 31;
    if (wave >= Tc) return;

    const float* xt = x + (size_t)wave * Hc;

    float sum = 0.f, sumsq = 0.f;
    float logit[Ec];
#pragma unroll
    for (int e = 0; e < Ec; ++e) logit[e] = 0.f;

#pragma unroll 4
    for (int i = 0; i < Hc / 32; ++i) {
        const int h = lane + 32 * i;
        const float v = xt[h];
        sum += v; sumsq += v * v;
        const float* wr = rw + (size_t)h * Ec;
#pragma unroll
        for (int e = 0; e < Ec; ++e) logit[e] += v * wr[e];
    }

#pragma unroll
    for (int off = 16; off > 0; off >>= 1) {
        sum   += __shfl_xor(sum,   off, 32);
        sumsq += __shfl_xor(sumsq, off, 32);
#pragma unroll
        for (int e = 0; e < Ec; ++e) logit[e] += __shfl_xor(logit[e], off, 32);
    }

    const float mu   = sum / (float)Hc;
    const float var  = sumsq / (float)Hc - mu * mu;
    const float rsig = rsqrtf(var + EPSc);

    float p[Ec];
    float mx = -1e30f;
#pragma unroll
    for (int e = 0; e < Ec; ++e) { logit[e] += rb[e]; mx = fmaxf(mx, logit[e]); }
    float den = 0.f;
#pragma unroll
    for (int e = 0; e < Ec; ++e) { p[e] = __expf(logit[e] - mx); den += p[e]; }
    const float rden = 1.f / den;
#pragma unroll
    for (int e = 0; e < Ec; ++e) p[e] *= rden;

    int i1 = 0;
#pragma unroll
    for (int e = 1; e < Ec; ++e) if (p[e] > p[i1]) i1 = e;
    int i2 = (i1 == 0) ? 1 : 0;
#pragma unroll
    for (int e = 0; e < Ec; ++e) if (e != i1 && p[e] > p[i2]) i2 = e;
    const float rs = 1.f / (p[i1] + p[i2]);

    if (lane < Ec) {
        score_out[(size_t)wave * Ec + lane] = p[lane];
        float cw = 0.f;
        if (lane == i1) cw = p[i1] * rs;
        if (lane == i2) cw = p[i2] * rs;
        combine[(size_t)wave * Ec + lane] = cw;
    }

    _Float16* xw = xn + (size_t)wave * Hc;
#pragma unroll 4
    for (int i = 0; i < Hc / 32; ++i) {
        const int h = lane + 32 * i;
        xw[h] = (_Float16)((xt[h] - mu) * rsig);
    }
}

// ---------------------------------------------------------------------
// Kernel 2: weight precondition. out[e][n][k] = (f16)(scale_k[e][k] * W[e][k][n])
// (scale_k == nullptr -> 1.0). 32x32 tile transpose through LDS.
// grid: (N/32, K/32, E)
// ---------------------------------------------------------------------
__global__ __launch_bounds__(256)
void transpose_f16_kernel(const float* __restrict__ Wbase,   // [E,K,N]
                          const float* __restrict__ scale,   // [E,K] or nullptr
                          _Float16* __restrict__ outbase,    // [E,N,K]
                          int K, int N)
{
    __shared__ _Float16 tile[32][33];
    const int e  = blockIdx.z;
    const int n0 = blockIdx.x * 32;
    const int k0 = blockIdx.y * 32;
    const int tx = threadIdx.x & 31;
    const int ty = threadIdx.x >> 5;   // 0..7

    const float* W = Wbase + (size_t)e * K * N;
    const float* sc = scale ? (scale + (size_t)e * K) : nullptr;
    _Float16* out = outbase + (size_t)e * N * K;

#pragma unroll
    for (int i = 0; i < 4; ++i) {
        const int r = ty * 4 + i;
        const float s = sc ? sc[k0 + r] : 1.f;
        tile[r][tx] = (_Float16)(s * W[(size_t)(k0 + r) * N + n0 + tx]);
    }
    __syncthreads();
#pragma unroll
    for (int i = 0; i < 4; ++i) {
        const int r = ty * 4 + i;
        out[(size_t)(n0 + r) * K + k0 + tx] = tile[tx][r];
    }
}

// ---------------------------------------------------------------------
// Kernel 3: fold LN beta through fc weights: bias2[e][n] = fc_b[e][n] + sum_k b[e][k]*W[e][k][n]
// grid: (I/256, E)
// ---------------------------------------------------------------------
__global__ __launch_bounds__(256)
void bias_fold_kernel(const float* __restrict__ ln_b,  // [E,H]
                      const float* __restrict__ fc_w,  // [E,H,I]
                      const float* __restrict__ fc_b,  // [E,I]
                      float* __restrict__ bias2)       // [E,I]
{
    const int e = blockIdx.y;
    const int n = blockIdx.x * 256 + threadIdx.x;
    const float* bb = ln_b + (size_t)e * Hc;
    const float* W  = fc_w + (size_t)e * Hc * Ic;
    float acc = fc_b[(size_t)e * Ic + n];
    for (int k = 0; k < Hc; ++k) acc += bb[k] * W[(size_t)k * Ic + n];
    bias2[(size_t)e * Ic + n] = acc;
}

// ---------------------------------------------------------------------
// Kernel 4: fc GEMM (per expert):  hid = gelu( xn @ W'fc + bias2 )
// A = xn[T,H] f16, B = wfc_t[e][I,H] f16 (K-contiguous). Double-buffered
// async-to-LDS staging, 8 waves, each wave owns 16x128 of a 128x128 tile.
// ---------------------------------------------------------------------
__global__ __launch_bounds__(256)
void fc_kernel(const _Float16* __restrict__ xn,     // [T,H]
               const _Float16* __restrict__ wfc_t,  // [E,I,H]
               const float* __restrict__ bias2,     // [E,I]
               _Float16* __restrict__ hid,          // [T,I]
               int e)
{
    __shared__ __align__(16) _Float16 smem[4 * TILE_ELEMS];

    const int tid  = threadIdx.x;
    const int lane = tid & 31;
    const int wv   = tid >> 5;
    const int n0   = blockIdx.x * BN;
    const int m0   = blockIdx.y * BM;

    const _Float16* Bt = wfc_t + (size_t)e * Ic * Hc;   // [I,H]

    v8f acc[8];
#pragma unroll
    for (int nt = 0; nt < 8; ++nt)
#pragma unroll
        for (int r = 0; r < 8; ++r) acc[nt][r] = 0.f;

    // prologue: stage k0 = 0 into buffer 0
    stage_tile_async(smem + 0 * TILE_ELEMS, xn, Hc, m0, 0, tid);
    stage_tile_async(smem + 2 * TILE_ELEMS, Bt, Hc, n0, 0, tid);

    int cur = 0;
    for (int k0 = 0; k0 < Hc; k0 += BK) {
        wait_async0();       // this wave's copies for tile k0 done
        __syncthreads();     // all waves' copies done

        _Float16* Acur = smem + cur * TILE_ELEMS;
        _Float16* Bcur = smem + (2 + cur) * TILE_ELEMS;

        if (k0 + BK < Hc) {  // stage next tile into the other buffer
            stage_tile_async(smem + (cur ^ 1) * TILE_ELEMS,       xn, Hc, m0, k0 + BK, tid);
            stage_tile_async(smem + (2 + (cur ^ 1)) * TILE_ELEMS, Bt, Hc, n0, k0 + BK, tid);
        }

        const v16h a = load_frag(Acur, wv * 16, lane);
#pragma unroll
        for (int nt = 0; nt < 8; ++nt) {
            const v16h b = load_frag(Bcur, nt * 16, lane);
            acc[nt] = __builtin_amdgcn_wmma_f32_16x16x32_f16(
                false, a, false, b, (short)0, acc[nt], false, false);
        }
        cur ^= 1;
    }

    // epilogue: bias + gelu, f16 out
    const int rbase = m0 + wv * 16 + ((lane >> 4) * 8);
    const int cl    = lane & 15;
    const float* fb = bias2 + (size_t)e * Ic;
#pragma unroll
    for (int nt = 0; nt < 8; ++nt) {
        const int col = n0 + nt * 16 + cl;
        const float bias = fb[col];
#pragma unroll
        for (int r = 0; r < 8; ++r) {
            const float v = gelu_tanh(acc[nt][r] + bias);
            hid[(size_t)(rbase + r) * Ic + col] = (_Float16)v;
        }
    }
}

// ---------------------------------------------------------------------
// Kernel 5: proj GEMM (per expert): out (+)= s_t * ( hid @ W'pj + proj_b )
// ---------------------------------------------------------------------
__global__ __launch_bounds__(256)
void proj_kernel(const _Float16* __restrict__ hid,    // [T,I]
                 const _Float16* __restrict__ wpj_t,  // [E,H,I]
                 const float* __restrict__ proj_b,    // [E,H]
                 const float* __restrict__ combine,   // [T,E]
                 float* __restrict__ out,             // [T,H]
                 int e, int accumulate)
{
    __shared__ __align__(16) _Float16 smem[4 * TILE_ELEMS];

    const int tid  = threadIdx.x;
    const int lane = tid & 31;
    const int wv   = tid >> 5;
    const int n0   = blockIdx.x * BN;
    const int m0   = blockIdx.y * BM;

    const _Float16* Bt = wpj_t + (size_t)e * Hc * Ic;   // [H,I]

    v8f acc[8];
#pragma unroll
    for (int nt = 0; nt < 8; ++nt)
#pragma unroll
        for (int r = 0; r < 8; ++r) acc[nt][r] = 0.f;

    stage_tile_async(smem + 0 * TILE_ELEMS, hid, Ic, m0, 0, tid);
    stage_tile_async(smem + 2 * TILE_ELEMS, Bt,  Ic, n0, 0, tid);

    int cur = 0;
    for (int k0 = 0; k0 < Ic; k0 += BK) {
        wait_async0();
        __syncthreads();

        _Float16* Acur = smem + cur * TILE_ELEMS;
        _Float16* Bcur = smem + (2 + cur) * TILE_ELEMS;

        if (k0 + BK < Ic) {
            stage_tile_async(smem + (cur ^ 1) * TILE_ELEMS,       hid, Ic, m0, k0 + BK, tid);
            stage_tile_async(smem + (2 + (cur ^ 1)) * TILE_ELEMS, Bt,  Ic, n0, k0 + BK, tid);
        }

        const v16h a = load_frag(Acur, wv * 16, lane);
#pragma unroll
        for (int nt = 0; nt < 8; ++nt) {
            const v16h b = load_frag(Bcur, nt * 16, lane);
            acc[nt] = __builtin_amdgcn_wmma_f32_16x16x32_f16(
                false, a, false, b, (short)0, acc[nt], false, false);
        }
        cur ^= 1;
    }

    const int rbase = m0 + wv * 16 + ((lane >> 4) * 8);
    const int cl    = lane & 15;
    const float* pb = proj_b + (size_t)e * Hc;
#pragma unroll
    for (int nt = 0; nt < 8; ++nt) {
        const int col = n0 + nt * 16 + cl;
        const float bias = pb[col];
#pragma unroll
        for (int r = 0; r < 8; ++r) {
            const int row = rbase + r;
            const float s = combine[(size_t)row * Ec + e];
            const float v = s * (acc[nt][r] + bias);
            float* dst = &out[(size_t)row * Hc + col];
            if (accumulate) *dst += v; else *dst = v;
        }
    }
}

// ---------------------------------------------------------------------
// launch
// ---------------------------------------------------------------------
extern "C" void kernel_launch(void* const* d_in, const int* in_sizes, int n_in,
                              void* d_out, int out_size, void* d_ws, size_t ws_size,
                              hipStream_t stream) {
    const float* x        = (const float*)d_in[0];
    const float* router_w = (const float*)d_in[1];
    const float* router_b = (const float*)d_in[2];
    const float* ln_g     = (const float*)d_in[3];
    const float* ln_b     = (const float*)d_in[4];
    const float* fc_w     = (const float*)d_in[5];
    const float* fc_b     = (const float*)d_in[6];
    const float* proj_w   = (const float*)d_in[7];
    const float* proj_b   = (const float*)d_in[8];

    float* out   = (float*)d_out;                  // [T,H]
    float* score = out + (size_t)Tc * Hc;          // [T,E]

    // workspace layout
    char* ws = (char*)d_ws;
    size_t off = 0;
    _Float16* xn    = (_Float16*)(ws + off); off += (size_t)Tc * Hc * 2;        // 32 MB
    _Float16* hid   = (_Float16*)(ws + off); off += (size_t)Tc * Ic * 2;        // 128 MB
    _Float16* wfc_t = (_Float16*)(ws + off); off += (size_t)Ec * Ic * Hc * 2;   // 64 MB
    _Float16* wpj_t = (_Float16*)(ws + off); off += (size_t)Ec * Hc * Ic * 2;   // 64 MB
    float*   combine = (float*)(ws + off);   off += (size_t)Tc * Ec * 4;        // 0.5 MB
    float*   bias2   = (float*)(ws + off);   off += (size_t)Ec * Ic * 4;        // 128 KB

    // 1) router + LN
    router_ln_kernel<<<Tc / 8, 256, 0, stream>>>(x, router_w, router_b,
                                                 score, combine, xn);

    // 2) precondition weights: fold gamma into fc weights, transpose both to
    //    K-contiguous f16; fold beta into fc bias.
    transpose_f16_kernel<<<dim3(Ic / 32, Hc / 32, Ec), 256, 0, stream>>>(
        fc_w, ln_g, wfc_t, Hc, Ic);
    transpose_f16_kernel<<<dim3(Hc / 32, Ic / 32, Ec), 256, 0, stream>>>(
        proj_w, nullptr, wpj_t, Ic, Hc);
    bias_fold_kernel<<<dim3(Ic / 256, Ec), 256, 0, stream>>>(ln_b, fc_w, fc_b, bias2);

    // 3) per-expert WMMA GEMMs with async-to-LDS double buffering
    const dim3 blk(256);
    const dim3 grid_fc(Ic / BN, Tc / BM);   // 32 x 128
    const dim3 grid_pj(Hc / BN, Tc / BM);   // 8  x 128
    for (int e = 0; e < Ec; ++e) {
        fc_kernel<<<grid_fc, blk, 0, stream>>>(xn, wfc_t, bias2, hid, e);
        proj_kernel<<<grid_pj, blk, 0, stream>>>(hid, wpj_t, proj_b, combine,
                                                 out, e, e > 0 ? 1 : 0);
    }
}
